// GraphSAGELayer_46033459479141
// MI455X (gfx1250) — matrix operand order repaired and checked by reading, not compiled, not run on hardware
//
#include <hip/hip_runtime.h>
#include <hip/hip_bf16.h>

// ---------------------------------------------------------------------------
// GraphSAGE / GCN layer for MI455X (gfx1250, wave32, WMMA).
//
//   agg[v]  = sum_{(u->v)} feat[u]            (edge scatter-add, L2-resident)
//   ah[v]   = agg[v] / max(deg[v],1)
//   out     = feat @ W1^T + b1 + ah @ W2^T + b2
//
// GEMM precision: fp32 operands are split into bf16 hi+lo; each product uses
// 3x v_wmma_f32_16x16x32_bf16 (hi*hi + lo*hi + hi*lo), dropping only the
// ~2^-16 lo*lo term -> fp32-class accuracy with tensor-core throughput.
// ---------------------------------------------------------------------------

typedef __attribute__((ext_vector_type(16))) __bf16          v16bf;
typedef __attribute__((ext_vector_type(8)))  float           v8f;
typedef __attribute__((ext_vector_type(16))) unsigned short  v16us;
typedef __attribute__((ext_vector_type(8)))  unsigned short  v8us;

__device__ __forceinline__ unsigned short f32_to_bf16_rne(float f) {
    unsigned int u = __builtin_bit_cast(unsigned int, f);
    unsigned int r = u + 0x7fffu + ((u >> 16) & 1u);   // round-to-nearest-even
    return (unsigned short)(r >> 16);
}
__device__ __forceinline__ float bf16_to_f32(unsigned short h) {
    unsigned int u = ((unsigned int)h) << 16;
    return __builtin_bit_cast(float, u);
}

// -------------------------------- kernel 0: zero scratch -------------------
__global__ void __launch_bounds__(256) gcn_zero(float* __restrict__ p, long n) {
    long i = (long)blockIdx.x * blockDim.x + threadIdx.x;
    long stride = (long)gridDim.x * blockDim.x;
    for (; i < n; i += stride) p[i] = 0.0f;
}

// ------------------- kernel 1: split W1/W2 into bf16 hi/lo, fuse biases ----
__global__ void __launch_bounds__(256) gcn_prep(
    const float* __restrict__ W1, const float* __restrict__ b1,
    const float* __restrict__ W2, const float* __restrict__ b2,
    unsigned short* __restrict__ w1hi, unsigned short* __restrict__ w1lo,
    unsigned short* __restrict__ w2hi, unsigned short* __restrict__ w2lo,
    float* __restrict__ bsum)
{
    int idx = blockIdx.x * 256 + threadIdx.x;          // 0..32767
    const float*     W   = (idx < 16384) ? W1   : W2;
    unsigned short*  Whi = (idx < 16384) ? w1hi : w2hi;
    unsigned short*  Wlo = (idx < 16384) ? w1lo : w2lo;
    int j = idx & 16383;
    float w = W[j];
    unsigned short h = f32_to_bf16_rne(w);
    Whi[j] = h;
    Wlo[j] = f32_to_bf16_rne(w - bf16_to_f32(h));
    if (idx < 128) bsum[idx] = b1[idx] + b2[idx];
}

// ------------------- kernel 2: edge aggregation (one wave per edge) --------
// Gather one feat row (float4/lane) and scatter-add into agg. Both arrays are
// L2-resident (25.6 MB each vs 192 MB L2). NOTE: float atomics => sum order
// is nondeterministic bitwise (inherent to parallel scatter-add).
__global__ void __launch_bounds__(256) gcn_aggregate(
    const float* __restrict__ feat, const int* __restrict__ src,
    const int* __restrict__ dst, float* __restrict__ agg,
    float* __restrict__ deg, int nEdges)
{
    int e = blockIdx.x * 8 + (threadIdx.x >> 5);
    if (e >= nEdges) return;
    int lane = threadIdx.x & 31;
    int s = src[e];
    int d = dst[e];
    const float4 v = *(const float4*)(feat + (size_t)s * 128 + lane * 4);
    float* ap = agg + (size_t)d * 128 + lane * 4;
    atomicAdd(ap + 0, v.x);
    atomicAdd(ap + 1, v.y);
    atomicAdd(ap + 2, v.z);
    atomicAdd(ap + 3, v.w);
    if (lane == 0) atomicAdd(deg + d, 1.0f);
}

// ------------------- WMMA fragment loaders (wave32 ISA layouts) ------------
// A (16x32 bf16): lane<16 -> row M=lane, a[0..7]=K k0..k0+7, a[8..15]=K k0+16..23
//                 lane>=16 -> row M=lane-16, K shifted by +8 / +24.
__device__ __forceinline__ v16bf load_a_frag(const unsigned short* base, // [16][136]
                                             int lane, int k0)
{
    int row   = lane & 15;
    int kbase = k0 + ((lane & 16) ? 8 : 0);
    const unsigned short* p = base + row * 136 + kbase;
    v8us x0 = *(const v8us*)(p);        // K = kbase .. kbase+7
    v8us x1 = *(const v8us*)(p + 16);   // K = kbase+16 .. kbase+23
    v16us a;
#pragma unroll
    for (int i = 0; i < 8; ++i) { a[i] = x0[i]; a[8 + i] = x1[i]; }
    return __builtin_bit_cast(v16bf, a);
}

// B (32x16 bf16), B[k][n] = W[n][k]: lane<16 -> col N=n0+lane, b[0..7]=K k0..k0+7, b[8..15]=K k0+16..23;
// lane>=16 -> col N=n0+lane-16, K shifted by +8 / +24. W is row-major [128][128] bf16.
__device__ __forceinline__ v16bf load_b_frag(const unsigned short* __restrict__ W,
                                             int n0, int lane, int k0)
{
    int n     = n0 + (lane & 15);
    int kbase = k0 + ((lane & 16) ? 8 : 0);
    const unsigned short* p = W + (size_t)n * 128 + kbase;
    v8us x0 = *(const v8us*)(p);
    v8us x1 = *(const v8us*)(p + 16);
    v16us b;
#pragma unroll
    for (int i = 0; i < 8; ++i) { b[i] = x0[i]; b[8 + i] = x1[i]; }
    return __builtin_bit_cast(v16bf, b);
}

__device__ __forceinline__ v8f wmma_bf16(v16bf a, v16bf b, v8f c) {
    return __builtin_amdgcn_wmma_f32_16x16x32_bf16(
        false, a, false, b, (short)0, c, false, false);
}

// ------------------- kernel 3: normalize + dual GEMM + bias ----------------
// Block = 16 nodes x 128 out-features. 8 waves, each owns one 16x16 N-tile.
// K = 128 in 4 chunks of 32 -> 6 WMMAs/chunk (hi/lo split, two matrices)
// -> 24 v_wmma_f32_16x16x32_bf16 per wave.
__global__ void __launch_bounds__(256) gcn_gemm(
    const float* __restrict__ feat, const float* __restrict__ agg,
    const float* __restrict__ deg,
    const unsigned short* __restrict__ w1hi, const unsigned short* __restrict__ w1lo,
    const unsigned short* __restrict__ w2hi, const unsigned short* __restrict__ w2lo,
    const float* __restrict__ bsum, float* __restrict__ out, int nNodes)
{
    // [0]=feat_hi [1]=feat_lo [2]=ah_hi [3]=ah_lo ; rows padded 128->136
    // (272 B row stride) to spread the 64 LDS banks.
    __shared__ unsigned short sA[4][16][136];

    const int tid   = threadIdx.x;
    const int node0 = blockIdx.x * 16;
    const int wv    = tid >> 5;
    const int lane  = tid & 31;
    const int n0    = wv * 16;

    // Warm L2/WGP$ for this wave's W rows (gfx1250 global_prefetch_b8).
    __builtin_prefetch(w1hi + (size_t)(n0 + (lane & 15)) * 128, 0, 3);
    __builtin_prefetch(w2hi + (size_t)(n0 + (lane & 15)) * 128, 0, 3);

    // ---- stage A tiles: each thread handles 8 columns of one row ----------
    {
        int r = tid >> 4;              // 0..15
        int c = (tid & 15) * 8;        // 0..120
        int node = node0 + r;
        float fv[8], av[8];
        if (node < nNodes) {
            const float* fp = feat + (size_t)node * 128 + c;
            float4 f0 = *(const float4*)(fp);
            float4 f1 = *(const float4*)(fp + 4);
            const float* ap = agg + (size_t)node * 128 + c;
            float4 a0 = *(const float4*)(ap);
            float4 a1 = *(const float4*)(ap + 4);
            float inv = 1.0f / fmaxf(deg[node], 1.0f);
            fv[0]=f0.x; fv[1]=f0.y; fv[2]=f0.z; fv[3]=f0.w;
            fv[4]=f1.x; fv[5]=f1.y; fv[6]=f1.z; fv[7]=f1.w;
            av[0]=a0.x*inv; av[1]=a0.y*inv; av[2]=a0.z*inv; av[3]=a0.w*inv;
            av[4]=a1.x*inv; av[5]=a1.y*inv; av[6]=a1.z*inv; av[7]=a1.w*inv;
        } else {
#pragma unroll
            for (int i = 0; i < 8; ++i) { fv[i] = 0.0f; av[i] = 0.0f; }
        }
#pragma unroll
        for (int i = 0; i < 8; ++i) {
            unsigned short h = f32_to_bf16_rne(fv[i]);
            sA[0][r][c + i] = h;
            sA[1][r][c + i] = f32_to_bf16_rne(fv[i] - bf16_to_f32(h));
            unsigned short g = f32_to_bf16_rne(av[i]);
            sA[2][r][c + i] = g;
            sA[3][r][c + i] = f32_to_bf16_rne(av[i] - bf16_to_f32(g));
        }
    }
    __syncthreads();

    // ---- WMMA main loop ---------------------------------------------------
    v8f acc = {};
#pragma unroll
    for (int kc = 0; kc < 4; ++kc) {
        int k0 = kc * 32;
        v16bf afh = load_a_frag(&sA[0][0][0], lane, k0);
        v16bf afl = load_a_frag(&sA[1][0][0], lane, k0);
        v16bf agh = load_a_frag(&sA[2][0][0], lane, k0);
        v16bf agl = load_a_frag(&sA[3][0][0], lane, k0);
        v16bf b1h = load_b_frag(w1hi, n0, lane, k0);
        v16bf b1l = load_b_frag(w1lo, n0, lane, k0);
        v16bf b2h = load_b_frag(w2hi, n0, lane, k0);
        v16bf b2l = load_b_frag(w2lo, n0, lane, k0);

        acc = wmma_bf16(afh, b1h, acc);   // feat_hi * W1_hi
        acc = wmma_bf16(afl, b1h, acc);   // feat_lo * W1_hi
        acc = wmma_bf16(afh, b1l, acc);   // feat_hi * W1_lo
        acc = wmma_bf16(agh, b2h, acc);   // ah_hi   * W2_hi
        acc = wmma_bf16(agl, b2h, acc);   // ah_lo   * W2_hi
        acc = wmma_bf16(agh, b2l, acc);   // ah_hi   * W2_lo
    }

    // ---- epilogue: C layout -> lane<16: c[r]=C[m=r][n=lane];
    //                             lane>=16: c[r]=C[m=8+r][n=lane-16]
    int n     = n0 + (lane & 15);
    int mbase = (lane & 16) ? 8 : 0;
    float bb  = bsum[n];
#pragma unroll
    for (int r = 0; r < 8; ++r) {
        int m = node0 + mbase + r;
        if (m < nNodes) out[(size_t)m * 128 + n] = acc[r] + bb;
    }
}

// ---------------------------------------------------------------------------
extern "C" void kernel_launch(void* const* d_in, const int* in_sizes, int n_in,
                              void* d_out, int out_size, void* d_ws, size_t ws_size,
                              hipStream_t stream) {
    const float* feat = (const float*)d_in[0];
    const int*   src  = (const int*)d_in[1];
    const int*   dst  = (const int*)d_in[2];
    const float* W1   = (const float*)d_in[3];
    const float* b1   = (const float*)d_in[4];
    const float* W2   = (const float*)d_in[5];
    const float* b2   = (const float*)d_in[6];
    float* out = (float*)d_out;

    const int nEdges = in_sizes[1];
    const int nNodes = in_sizes[0] / 128;   // 50000

    // workspace layout
    char*  ws       = (char*)d_ws;
    size_t aggBytes = (size_t)nNodes * 128 * sizeof(float);   // 25.6 MB
    size_t degBytes = (size_t)nNodes * sizeof(float);         // 200 KB
    float* agg = (float*)(ws);
    float* deg = (float*)(ws + aggBytes);
    size_t off = (aggBytes + degBytes + 255) & ~(size_t)255;
    unsigned short* w1hi = (unsigned short*)(ws + off); off += 128 * 128 * 2;
    unsigned short* w1lo = (unsigned short*)(ws + off); off += 128 * 128 * 2;
    unsigned short* w2hi = (unsigned short*)(ws + off); off += 128 * 128 * 2;
    unsigned short* w2lo = (unsigned short*)(ws + off); off += 128 * 128 * 2;
    float* bsum = (float*)(ws + off);

    // 0) zero agg+deg (contiguous)
    long nZero = (long)nNodes * 129;
    gcn_zero<<<2048, 256, 0, stream>>>(agg, nZero);

    // 1) weight split + bias fuse (exactly 32768 threads)
    gcn_prep<<<128, 256, 0, stream>>>(W1, b1, W2, b2, w1hi, w1lo, w2hi, w2lo, bsum);

    // 2) edge aggregation: one wave per edge
    gcn_aggregate<<<(nEdges + 7) / 8, 256, 0, stream>>>(feat, src, dst, agg, deg, nEdges);

    // 3) normalize + dual GEMM + bias: one block per 16 nodes
    gcn_gemm<<<(nNodes + 15) / 16, 256, 0, stream>>>(
        feat, agg, deg, w1hi, w1lo, w2hi, w2lo, bsum, out, nNodes);
}
